// ScaledDotProductAttention_7181185319548
// MI455X (gfx1250) — matrix operand order repaired and checked by reading, not compiled, not run on hardware
//
#include <hip/hip_runtime.h>
#include <hip/hip_bf16.h>

// ---------------------------------------------------------------------------
// Scaled dot-product attention for MI455X (gfx1250, wave32, WMMA).
//   attn = softmax(QK^T/8 + mask, masked_fill(mask<0, -1e30))
//   out  = attn @ V
// One 256-thread (8-wave) block per (batch, 16-row Q tile).
// Scores (16 x 2048 f32 = 128 KB) staged in LDS (CDNA5: 320 KB/WGP keeps
// 2 workgroups resident). HBM-bound on mask read + attn write (~2.25 GB).
// ---------------------------------------------------------------------------

typedef __attribute__((ext_vector_type(16))) __bf16 v16bf;
typedef __attribute__((ext_vector_type(8)))  float  v8f;

#define BSZ      64
#define LQ       2048
#define LK       2048
#define DH       64
#define QT       16          // query rows per workgroup
#define INV_TEMP 0.125f      // 1/8
#define NEG_INF  (-1e30f)

// Native conversion: lets the backend emit v_cvt_pk_bf16_f32 (packed, 2/instr)
// instead of a manual integer-rounding sequence.
__device__ __forceinline__ __bf16 f2bf(float f) { return (__bf16)f; }

__global__ __launch_bounds__(256)
void attn_fused_kernel(const float* __restrict__ q,
                       const float* __restrict__ k,
                       const float* __restrict__ v,
                       const float* __restrict__ mask,
                       float* __restrict__ out,
                       float* __restrict__ attn) {
  extern __shared__ char smem[];
  float*  sc   = (float*)smem;                                // [QT][LK]  128 KB
  __bf16* qt   = (__bf16*)(smem + (size_t)QT * LK * 4);       // [QT][DH]    2 KB
  float*  obuf = (float*)(smem + (size_t)QT * LK * 4
                               + (size_t)QT * DH * 2);        // [4][16][16] 4 KB

  const int b    = blockIdx.y;
  const int q0   = blockIdx.x * QT;
  const int tid  = threadIdx.x;
  const int wid  = tid >> 5;
  const int lane = tid & 31;
  const int half = lane >> 4;    // 0: lanes 0-15, 1: lanes 16-31
  const int n16  = lane & 15;

  // ------------------------------------------------------------------
  // Stage Q tile (16 x 64) into LDS as bf16
  // ------------------------------------------------------------------
  for (int i = tid; i < QT * DH; i += 256) {
    int r = i / DH, d = i % DH;
    qt[i] = f2bf(q[((size_t)b * LQ + q0 + r) * DH + d]);
  }
  __syncthreads();

  // A fragments for QK^T (16x32 bf16, contiguous k order shared with B), reused all tiles.
  // lane<16: M=lane, K = 0..15; lane>=16: M=lane-16, K = 16..31
  v16bf a0, a1;
#pragma unroll
  for (int e = 0; e < 16; ++e) {
    int kk = e + 16 * half;
    a0[e] = qt[n16 * DH + kk];        // d  0..31
    a1[e] = qt[n16 * DH + 32 + kk];   // d 32..63
  }

  const float* kbase = k + (size_t)b * LK * DH;
  const float* mrow  = mask + ((size_t)b * LQ + q0) * LK;

  // ------------------------------------------------------------------
  // Phase 1: S = QK^T/8 + mask (masked_fill), staged into LDS
  // 8 waves x 16 cols = 128 cols per sweep, 16 sweeps over LK
  // ------------------------------------------------------------------
  for (int c0 = wid * 16; c0 < LK; c0 += 128) {
    // B fragment: K-matrix rows (c0+n16), head-dims contiguous -> b128 loads.
    // lane<16: K=0..15 ; lane>=16: K=16..31 (two consecutive per VGPR)
    const float4* kp4 =
        (const float4*)(kbase + (size_t)(c0 + n16) * DH + 16 * half);
    v16bf b0, b1;
#pragma unroll
    for (int j = 0; j < 4; ++j) {
      float4 f = kp4[j];          // d-offset  0..31 block
      b0[4 * j + 0] = f2bf(f.x); b0[4 * j + 1] = f2bf(f.y);
      b0[4 * j + 2] = f2bf(f.z); b0[4 * j + 3] = f2bf(f.w);
      float4 g = kp4[j + 8];      // d-offset 32..63 block
      b1[4 * j + 0] = f2bf(g.x); b1[4 * j + 1] = f2bf(g.y);
      b1[4 * j + 2] = f2bf(g.z); b1[4 * j + 3] = f2bf(g.w);
    }

    v8f c = {};
    c = __builtin_amdgcn_wmma_f32_16x16x32_bf16(false, a0, false, b0,
                                                (short)0, c, false, false);
    c = __builtin_amdgcn_wmma_f32_16x16x32_bf16(false, a1, false, b1,
                                                (short)0, c, false, false);

    // Prefetch next sweep's mask tile (global_prefetch_b8)
    if (c0 + 128 < LK)
      __builtin_prefetch(mrow + (size_t)(8 * half) * LK + c0 + 128 + n16, 0, 1);

    // C layout: lane<16 -> M=r, N=lane ; lane>=16 -> M=r+8, N=lane-16
#pragma unroll
    for (int r = 0; r < 8; ++r) {
      int   m   = r + 8 * half;
      int   col = c0 + n16;
      float mv  = mrow[(size_t)m * LK + col];
      float s   = c[r] * INV_TEMP + mv;
      if (mv < 0.f) s = NEG_INF;
      sc[m * LK + col] = s;
    }
  }
  __syncthreads();

  // ------------------------------------------------------------------
  // Phase 2: softmax over each of the 16 rows (one row per 16-lane group)
  // ------------------------------------------------------------------
  {
    int    row  = wid * 2 + half;   // 8 waves x 2 halves = 16 rows
    float* srow = sc + row * LK;

    float mx = -3.4e38f;
    for (int cc = n16; cc < LK; cc += 16) mx = fmaxf(mx, srow[cc]);
#pragma unroll
    for (int o = 1; o < 16; o <<= 1) mx = fmaxf(mx, __shfl_xor(mx, o, 32));

    float sum = 0.f;
    for (int cc = n16; cc < LK; cc += 16) {
      float e = __expf(srow[cc] - mx);
      srow[cc] = e;
      sum += e;
    }
#pragma unroll
    for (int o = 1; o < 16; o <<= 1) sum += __shfl_xor(sum, o, 32);

    float inv = 1.f / sum;
    for (int cc = n16; cc < LK; cc += 16) srow[cc] *= inv;
  }
  __syncthreads();

  // Coalesced write of normalized attn block (rows are contiguous in global)
  {
    float4*       adst = (float4*)(attn + ((size_t)b * LQ + q0) * LK);
    const float4* asrc = (const float4*)sc;
    for (int i = tid; i < QT * LK / 4; i += 256) adst[i] = asrc[i];
  }

  // ------------------------------------------------------------------
  // Phase 3: out = P @ V.  8 waves = 4 N-tiles x 2 K-halves of 1024.
  // ------------------------------------------------------------------
  const int    ntile = wid & 3;      // 16-wide slice of the 64 head dims
  const int    kpart = wid >> 2;     // which 1024-column half of LK
  const float* vb    = v + (size_t)b * LK * DH;

  v8f oc = {};
  for (int kb = kpart * (LK / 2); kb < (kpart + 1) * (LK / 2); kb += 32) {
    v16bf pa, vf;
#pragma unroll
    for (int e = 0; e < 16; ++e) {
      int kk = e + 16 * half;
      pa[e] = f2bf(sc[n16 * LK + kb + kk]);          // A = probabilities
    }
#pragma unroll
    for (int e = 0; e < 16; ++e) {
      int kr = kb + e + 16 * half;                    // B = V rows
      vf[e] = f2bf(vb[(size_t)kr * DH + ntile * 16 + n16]);
    }
    oc = __builtin_amdgcn_wmma_f32_16x16x32_bf16(false, pa, false, vf,
                                                 (short)0, oc, false, false);
  }

  if (kpart == 1) {
#pragma unroll
    for (int r = 0; r < 8; ++r)
      obuf[(ntile * 16 + (r + 8 * half)) * 16 + n16] = oc[r];
  }
  __syncthreads();
  if (kpart == 0) {
    float* od = out + ((size_t)b * LQ + q0) * DH;
#pragma unroll
    for (int r = 0; r < 8; ++r) {
      int m = r + 8 * half;
      od[(size_t)m * DH + ntile * 16 + n16] =
          oc[r] + obuf[(ntile * 16 + m) * 16 + n16];
    }
  }
}

extern "C" void kernel_launch(void* const* d_in, const int* in_sizes, int n_in,
                              void* d_out, int out_size, void* d_ws,
                              size_t ws_size, hipStream_t stream) {
  (void)n_in; (void)out_size; (void)d_ws; (void)ws_size;
  const float* q    = (const float*)d_in[0];
  const float* k    = (const float*)d_in[1];
  const float* v    = (const float*)d_in[2];
  const float* mask = (const float*)d_in[3];

  const int bsz = in_sizes[0] / (LQ * DH);   // 64

  float* out  = (float*)d_out;
  float* attn = out + (size_t)bsz * LQ * DH; // outputs concatenated: (out, attn)

  size_t shmem = (size_t)QT * LK * 4   // scores
               + (size_t)QT * DH * 2   // bf16 Q tile
               + 4 * 16 * 16 * 4;      // PV partial reduction
  dim3 grid(LQ / QT, bsz);
  attn_fused_kernel<<<grid, 256, shmem, stream>>>(q, k, v, mask, out, attn);
}
